// RTGU_27487790694822
// MI455X (gfx1250) — 8-bit matrix operand order repaired and checked by reading, compiled, not run on hardware
//
#include <hip/hip_runtime.h>


typedef int   v8i __attribute__((ext_vector_type(8)));
typedef float v8f __attribute__((ext_vector_type(8)));

#define TT   4096
#define HID  256
#define GATE_BYTES (8*16*1024)

#define WS_WPREP 0
#define WS_XBUF  786432
#define WS_FLAGS 819200

#define L_WRZ 0
#define L_A   262144
#define L_RH  270336
#define L_HF  274432
#define L_TOT 290816

__device__ __forceinline__ unsigned int cvt1_fp8(float a) {
  return (unsigned int)__builtin_amdgcn_cvt_pk_fp8_f32(a, a, 0, false) & 0xFFu;
}
__device__ __forceinline__ unsigned int cvt4_fp8(float a, float b, float c, float d) {
  int v = __builtin_amdgcn_cvt_pk_fp8_f32(a, b, 0, false);
  v = __builtin_amdgcn_cvt_pk_fp8_f32(c, d, v, true);
  return (unsigned int)v;
}

__global__ void gru_prep_weights(const float* __restrict__ Wr,
                                 const float* __restrict__ Wz,
                                 const float* __restrict__ Wh,
                                 unsigned char* __restrict__ wp) {
  int idx = blockIdx.x * blockDim.x + threadIdx.x;
  const int per = 512 * 256;
  if (idx >= 2 * 3 * per) return;
  int l = idx / (3 * per);
  int r = idx - l * 3 * per;
  int g = r / per;
  r -= g * per;
  int k = r >> 8;
  int n = r & 255;
  const float* W = (g == 0) ? Wr : ((g == 1) ? Wz : Wh);
  float w = W[(size_t)(l * 512 + k) * 256 + n];
  int kt = k >> 6, k64 = k & 63;
  int wt = n >> 4, n16 = n & 15;
  int lane = n16 + ((k64 & 16) ? 16 : 0);
  int vreg = ((k64 & 32) ? 4 : 0) + ((k64 & 12) >> 2);
  int byt  = k64 & 3;
  size_t dst = ((((size_t)(l * 3 + g) * 8 + kt) * 16 + wt) * 1024) + lane * 32 + vreg * 4 + byt;
  wp[dst] = (unsigned char)cvt1_fp8(w);
}

__global__ void gru_init_flags(unsigned int* flags) {
  if (threadIdx.x < 8) flags[threadIdx.x] = 0u;
}

__device__ __forceinline__ v8i load_afrag(const unsigned char* rowp) {
  uint2 q0 = *(const uint2*)(rowp);
  uint2 q1 = *(const uint2*)(rowp + 8);
  uint2 q2 = *(const uint2*)(rowp + 32);
  uint2 q3 = *(const uint2*)(rowp + 40);
  v8i a;
  a[0] = (int)q0.x; a[1] = (int)q0.y;
  a[2] = (int)q1.x; a[3] = (int)q1.y;
  a[4] = (int)q2.x; a[5] = (int)q2.y;
  a[6] = (int)q3.x; a[7] = (int)q3.y;
  return a;
}

__device__ __forceinline__ float sigmoidf_fast(float x) {
  return 1.f / (1.f + __expf(-x));
}
__device__ __forceinline__ float tanhf_fast(float x) {
  return 2.f / (1.f + __expf(-2.f * x)) - 1.f;
}

__global__ void __launch_bounds__(512, 1)
gru_main(const int* __restrict__ tokens, const float* __restrict__ emb,
         const float* __restrict__ brv, const float* __restrict__ bzv,
         const float* __restrict__ bhv, const float* __restrict__ Wfc,
         const float* __restrict__ bfc, const unsigned char* __restrict__ wp,
         unsigned char* __restrict__ xbuf, unsigned int* __restrict__ flags,
         float* __restrict__ out) {
  const int tid   = threadIdx.x;
  const int lane  = tid & 31;
  const int wv    = tid >> 5;
  const int tile  = blockIdx.x & 3;
  const int layer = blockIdx.x >> 2;

  extern __shared__ unsigned char smem[];
  unsigned char* Wlds = smem + L_WRZ;
  unsigned char* Abuf = smem + L_A;
  unsigned char* RHb  = smem + L_RH;
  float*         HF   = (float*)(smem + L_HF);

  {
    const int4* s4 = (const int4*)(wp + (size_t)(layer * 3) * GATE_BYTES);
    int4* d4 = (int4*)Wlds;
    for (int i = tid; i < 262144 / 16; i += 512) d4[i] = s4[i];
  }
  v8i wh[8];
  {
    const unsigned char* src = wp + (size_t)(layer * 3 + 2) * GATE_BYTES;
#pragma unroll
    for (int kt = 0; kt < 8; ++kt)
      wh[kt] = *(const v8i*)(src + ((size_t)(kt * 16 + wv)) * 1024 + lane * 32);
  }
  for (int i = tid; i < 16 * 64; i += 512) {
    int row = i >> 6, c4 = i & 63;
    *(unsigned int*)(Abuf + row * 512 + 256 + c4 * 4) = 0u;
  }

  const int c_loc  = (wv << 4) | (lane & 15);
  const int rowoff = (lane >> 4) << 3;
  const int m16    = lane & 15;
  const int khalf  = (lane >> 4) << 4;
  const float bias_r = brv[layer * HID + c_loc];
  const float bias_z = bzv[layer * HID + c_loc];
  const float bias_h = bhv[layer * HID + c_loc];

  float hst[8];
#pragma unroll
  for (int i = 0; i < 8; ++i) hst[i] = 0.f;

  unsigned int* prodF = flags + tile * 2 + 0;
  unsigned int* consF = flags + tile * 2 + 1;
  unsigned char* xb   = xbuf + (size_t)tile * 2 * 4096;

  for (int t = 0; t < TT; ++t) {
    __syncthreads();
    if (layer == 0) {
      if (t >= 2) {
        if (tid == 0)
          while (__hip_atomic_load(consF, __ATOMIC_ACQUIRE,
                                   __HIP_MEMORY_SCOPE_AGENT) < (unsigned)(t - 1))
            __builtin_amdgcn_s_sleep(1);
        __syncthreads();
      }
      int row = tid >> 5;
      int e0  = lane << 3;
      int tok = tokens[(size_t)(tile * 16 + row) * TT + t];
      if (t + 1 < TT)
        __builtin_prefetch(&tokens[(size_t)(tile * 16 + row) * TT + t + 1], 0, 0);
      const float* ep = emb + (size_t)tok * HID + e0;
      float4 x0 = *(const float4*)(ep);
      float4 x1 = *(const float4*)(ep + 4);
      *(unsigned int*)(Abuf + row * 512 + e0)     = cvt4_fp8(x0.x, x0.y, x0.z, x0.w);
      *(unsigned int*)(Abuf + row * 512 + e0 + 4) = cvt4_fp8(x1.x, x1.y, x1.z, x1.w);
    } else {
      if (tid == 0)
        while (__hip_atomic_load(prodF, __ATOMIC_ACQUIRE,
                                 __HIP_MEMORY_SCOPE_AGENT) < (unsigned)(t + 1))
          __builtin_amdgcn_s_sleep(1);
      __syncthreads();
      const unsigned char* src = xb + (t & 1) * 4096;
      int row = tid >> 5;
      int off = lane << 3;
      *(uint2*)(Abuf + row * 512 + off) = *(const uint2*)(src + row * 256 + off);
      __syncthreads();
      if (tid == 0)
        __hip_atomic_store(consF, (unsigned)(t + 1), __ATOMIC_RELEASE,
                           __HIP_MEMORY_SCOPE_AGENT);
    }
    __syncthreads();

    v8f accr = {0.f, 0.f, 0.f, 0.f, 0.f, 0.f, 0.f, 0.f};
    v8f accz = {0.f, 0.f, 0.f, 0.f, 0.f, 0.f, 0.f, 0.f};
#pragma unroll
    for (int kt = 0; kt < 8; ++kt) {
      v8i af = load_afrag(Abuf + m16 * 512 + kt * 64 + khalf);
      v8i br_ = *(const v8i*)(Wlds + (size_t)(kt * 16 + wv) * 1024 + lane * 32);
      v8i bz_ = *(const v8i*)(Wlds + 131072 + (size_t)(kt * 16 + wv) * 1024 + lane * 32);
      accr = __builtin_amdgcn_wmma_f32_16x16x64_fp8_fp8(af, br_, (short)0, accr, false, false);
      accz = __builtin_amdgcn_wmma_f32_16x16x64_fp8_fp8(af, bz_, (short)0, accz, false, false);
    }
    float zv[8];
#pragma unroll
    for (int i = 0; i < 8; ++i) {
      float rv = sigmoidf_fast(accr[i] + bias_r);
      zv[i]    = sigmoidf_fast(accz[i] + bias_z);
      RHb[(i + rowoff) * 256 + c_loc] = (unsigned char)cvt1_fp8(rv * hst[i]);
    }
    __syncthreads();

    v8f acch = {0.f, 0.f, 0.f, 0.f, 0.f, 0.f, 0.f, 0.f};
#pragma unroll
    for (int kt = 0; kt < 8; ++kt) {
      const unsigned char* rp = (kt < 4)
          ? (Abuf + m16 * 512 + kt * 64 + khalf)
          : (RHb  + m16 * 256 + (kt - 4) * 64 + khalf);
      v8i af = load_afrag(rp);
      acch = __builtin_amdgcn_wmma_f32_16x16x64_fp8_fp8(af, wh[kt], (short)0, acch, false, false);
    }
#pragma unroll
    for (int i = 0; i < 8; ++i) {
      float ht = tanhf_fast(acch[i] + bias_h);
      float hn = hst[i] + zv[i] * (ht - hst[i]);
      hst[i] = hn;
      unsigned char h8 = (unsigned char)cvt1_fp8(hn);
      Abuf[(i + rowoff) * 512 + 256 + c_loc] = h8;
      if (layer == 0) xb[(t & 1) * 4096 + (i + rowoff) * 256 + c_loc] = h8;
    }
    if (layer == 0) {
      __threadfence();
      __syncthreads();
      if (tid == 0)
        __hip_atomic_store(prodF, (unsigned)(t + 1), __ATOMIC_RELEASE,
                           __HIP_MEMORY_SCOPE_AGENT);
    }
  }

  if (layer == 1) {
#pragma unroll
    for (int i = 0; i < 8; ++i) HF[(i + rowoff) * 256 + c_loc] = hst[i];
    __syncthreads();
    if (tid < 32) {
      int m = tid >> 1, cls = tid & 1;
      float s = bfc[cls];
      for (int j = 0; j < 256; ++j) s += HF[m * 256 + j] * Wfc[j * 2 + cls];
      out[(tile * 16 + m) * 2 + cls] = s;
    }
  }
}

extern "C" void kernel_launch(void* const* d_in, const int* in_sizes, int n_in,
                              void* d_out, int out_size, void* d_ws, size_t ws_size,
                              hipStream_t stream) {
  (void)in_sizes; (void)n_in; (void)out_size; (void)ws_size;
  const int*   tokens = (const int*)d_in[0];
  const float* emb = (const float*)d_in[1];
  const float* Wr  = (const float*)d_in[2];
  const float* br  = (const float*)d_in[3];
  const float* Wz  = (const float*)d_in[4];
  const float* bz  = (const float*)d_in[5];
  const float* Wh  = (const float*)d_in[6];
  const float* bh  = (const float*)d_in[7];
  const float* Wfc = (const float*)d_in[8];
  const float* bfc = (const float*)d_in[9];

  unsigned char* wp    = (unsigned char*)d_ws + WS_WPREP;
  unsigned char* xbuf  = (unsigned char*)d_ws + WS_XBUF;
  unsigned int*  flags = (unsigned int*)((unsigned char*)d_ws + WS_FLAGS);
  float* out = (float*)d_out;

  gru_prep_weights<<<dim3((2 * 3 * 512 * 256) / 256), dim3(256), 0, stream>>>(Wr, Wz, Wh, wp);
  gru_init_flags<<<dim3(1), dim3(32), 0, stream>>>(flags);
  gru_main<<<dim3(8), dim3(512), L_TOT, stream>>>(tokens, emb, br, bz, bh,
                                                  Wfc, bfc, wp, xbuf, flags, out);
}
